// LocalCorrelation1D_13838384628361
// MI455X (gfx1250) — matrix operand order repaired and checked by reading, not compiled, not run on hardware
//
#include <hip/hip_runtime.h>
#include <hip/hip_fp16.h>
#include <math.h>

typedef __attribute__((ext_vector_type(16))) _Float16 v16h;
typedef __attribute__((ext_vector_type(8)))  float    v8f;
typedef __attribute__((ext_vector_type(4)))  unsigned  v4u;

namespace {
constexpr int B_ = 4, C_ = 128, H_ = 128, W_ = 240, E_ = 64;
constexpr int R_ = 4, K_ = 2 * R_ + 1;              // 9 deltas
constexpr int PHI_PITCH  = 136;  // halves per phi row: 128 + pad; 272 B row stride -> 16B aligned, bank stride 4
constexpr int FEAT_PITCH = 136;  // halves per feature pixel row (same properties)
constexpr int EMB_PITCH  = 66;   // floats per embedded pixel: conflict-free scalar reads in correlation
constexpr int PHI_BYTES  = E_ * PHI_PITCH * 2;       // 17408  (16B aligned)
constexpr int FEAT_BYTES = W_ * FEAT_PITCH * 2;      // 65280  (16B aligned)
constexpr int EMB_BYTES  = W_ * EMB_PITCH * 4;       // 63360
constexpr int SMEM_BYTES = PHI_BYTES + FEAT_BYTES + 2 * EMB_BYTES; // 209408 < 320 KB
}

union Frag { v4u q[2]; v16h h; };

// A-fragment: 16x32 f16 tile of phi (M = e rows, K = c).
// Lane (m=lane&15, hi=lane>>4) holds K = {hi*8 + 0..7} in V0..3 and {16 + hi*8 + 0..7} in V4..7,
// i.e. u32 row indices kbase/2 + hi*4 + {0..3} and kbase/2 + 8 + hi*4 + {0..3}: two b128 chunks.
__device__ inline v16h load_frag_A(const unsigned* phi32, int etile, int kbase, int lane) {
  const int m = lane & 15, hi = lane >> 4;
  const unsigned* p = phi32 + (etile * 16 + m) * (PHI_PITCH / 2) + (kbase >> 1) + hi * 4;
  Frag f;
  f.q[0] = *(const v4u*)(p);
  f.q[1] = *(const v4u*)(p + 8);
  return f.h;
}

// B-fragment: 32x16 f16 tile of features (K = c rows, N = pixel cols).
// Lane (n=lane&15, hi=lane>>4) holds K = {hi*8 + 0..7} and {16 + hi*8 + 0..7}: u32 row indices
// kbase/2 + hi*4 + {0..3} and kbase/2 + 8 + hi*4 + {0..3}: two b128 chunks.
__device__ inline v16h load_frag_B(const unsigned* feat32, int ptile, int kbase, int lane) {
  const int n = lane & 15, hi = lane >> 4;
  const unsigned* p = feat32 + (ptile * 16 + n) * (FEAT_PITCH / 2) + (kbase >> 1) + hi * 4;
  Frag f;
  f.q[0] = *(const v4u*)(p);
  f.q[1] = *(const v4u*)(p + 8);
  return f.h;
}

// Embed one full row: emb[pixel][e] = phi(64x128) @ feat(128x240) via v_wmma_f32_16x16x32_f16.
// job = wave + 8k  =>  etile = wave & 3 is wave-invariant: compiler hoists all 4 A-fragments.
__device__ inline void embed_row(const unsigned* phi32, const unsigned* feat32,
                                 float* emb, int wave, int lane) {
  const int n = lane & 15, hi = lane >> 4;
  for (int job = wave; job < (W_ / 16) * (E_ / 16); job += 8) {  // 60 jobs / 8 waves (wave-uniform)
    const int ptile = job >> 2, etile = job & 3;
    v8f acc = {};
#pragma unroll
    for (int ks = 0; ks < C_ / 32; ++ks) {
      v16h a = load_frag_A(phi32, etile, ks * 32, lane);
      v16h b = load_frag_B(feat32, ptile, ks * 32, lane);
      acc = __builtin_amdgcn_wmma_f32_16x16x32_f16(false, a, false, b,
                                                   (short)0, acc, false, false);
    }
    // D layout: VGPR r -> M = r + 8*hi (e index), N = lane&15 (pixel)
    float* dst = emb + (ptile * 16 + n) * EMB_PITCH + etile * 16 + 8 * hi;
#pragma unroll
    for (int r = 0; r < 8; ++r) dst[r] = acc[r];
  }
}

__device__ inline void normalize_vec(float* v) {
  float s = 0.f;
#pragma unroll
  for (int e = 0; e < E_; ++e) s += v[e] * v[e];
  const float inv = 1.0f / (sqrtf(s) + 1e-6f);
#pragma unroll
  for (int e = 0; e < E_; ++e) v[e] *= inv;
}

__global__ __launch_bounds__(256, 1)
void corr1d_fused_kernel(const float* __restrict__ FL, const float* __restrict__ FR,
                         const float* __restrict__ d0, const float* __restrict__ phiW,
                         float* __restrict__ out) {
  extern __shared__ char smem[];
  _Float16* phiH = (_Float16*)smem;
  _Float16* feat = (_Float16*)(smem + PHI_BYTES);
  float* embL = (float*)(smem + PHI_BYTES + FEAT_BYTES);
  float* embR = (float*)(smem + PHI_BYTES + FEAT_BYTES + EMB_BYTES);
  const unsigned* phi32  = (const unsigned*)smem;
  const unsigned* feat32 = (const unsigned*)(smem + PHI_BYTES);

  const int tid  = threadIdx.x;
  const int wave = tid >> 5, lane = tid & 31;
  const int b = blockIdx.x / H_, h = blockIdx.x % H_;

  // ---- stage phi (f16) ----
  for (int i = tid; i < E_ * C_; i += 256) {
    const int e = i >> 7, c = i & 127;
    phiH[e * PHI_PITCH + c] = (_Float16)phiW[i];
  }
  // ---- stage FL row (f16, pixel-major) ----
  const size_t rowBase = ((size_t)b * C_ * H_ + h) * W_;  // element (b,c,h,x) = rowBase + c*H*W + x
  for (int i = tid; i < C_ * W_; i += 256) {
    const int c = i / W_, x = i % W_;
    feat[x * FEAT_PITCH + c] = (_Float16)FL[rowBase + (size_t)c * H_ * W_ + x];
  }
  __syncthreads();

  // ---- embed left ----
  embed_row(phi32, feat32, embL, wave, lane);
  __syncthreads();

  // ---- stage FR row (reuse buffer) + normalize embL ----
  for (int i = tid; i < C_ * W_; i += 256) {
    const int c = i / W_, x = i % W_;
    feat[x * FEAT_PITCH + c] = (_Float16)FR[rowBase + (size_t)c * H_ * W_ + x];
  }
  if (tid < W_) normalize_vec(embL + tid * EMB_PITCH);
  __syncthreads();

  // ---- embed right ----
  embed_row(phi32, feat32, embR, wave, lane);
  __syncthreads();
  if (tid < W_) normalize_vec(embR + tid * EMB_PITCH);
  __syncthreads();

  // ---- correlation: 9 deltas x 240 pixels, linear interpolation along W ----
  const float* d0row = d0 + ((size_t)b * H_ + h) * W_;
  for (int item = tid; item < K_ * W_; item += 256) {
    const int k = item / W_, x = item % W_;
    const float dp  = d0row[x];
    const float xp  = (float)x - dp - (float)(k - R_);
    const float x0f = floorf(xp);
    const float w   = xp - x0f;
    int i0 = (int)x0f;
    int i1 = i0 + 1;
    i0 = i0 < 0 ? 0 : (i0 > W_ - 1 ? W_ - 1 : i0);
    i1 = i1 < 0 ? 0 : (i1 > W_ - 1 ? W_ - 1 : i1);
    const float* fl = embL + x * EMB_PITCH;
    const float* g0 = embR + i0 * EMB_PITCH;
    const float* g1 = embR + i1 * EMB_PITCH;
    float acc = 0.f;
#pragma unroll
    for (int e = 0; e < E_; ++e)
      acc = fmaf(fl[e], fmaf(w, g1[e] - g0[e], g0[e]), acc);
    out[(((size_t)b * K_ + k) * H_ + h) * W_ + x] = acc;
  }
}

extern "C" void kernel_launch(void* const* d_in, const int* in_sizes, int n_in,
                              void* d_out, int out_size, void* d_ws, size_t ws_size,
                              hipStream_t stream) {
  const float* FL   = (const float*)d_in[0];
  const float* FR   = (const float*)d_in[1];
  const float* d0   = (const float*)d_in[2];
  const float* phiW = (const float*)d_in[3];
  float* out = (float*)d_out;
  (void)in_sizes; (void)n_in; (void)out_size; (void)d_ws; (void)ws_size;

  hipFuncSetAttribute((const void*)corr1d_fused_kernel,
                      hipFuncAttributeMaxDynamicSharedMemorySize, SMEM_BYTES);
  corr1d_fused_kernel<<<dim3(B_ * H_), dim3(256), SMEM_BYTES, stream>>>(FL, FR, d0, phiW, out);
}
